// StrongOrthoPhasor_37082747633865
// MI455X (gfx1250) — matrix operand order repaired and checked by reading, not compiled, not run on hardware
//
#include <hip/hip_runtime.h>
#include <hip/hip_bf16.h>
#include <math.h>

typedef long long i64;
typedef __attribute__((ext_vector_type(16))) __bf16 v16bf;
typedef __attribute__((ext_vector_type(8)))  __bf16 v8bf;
typedef __attribute__((ext_vector_type(8)))  float  v8f;
typedef __attribute__((ext_vector_type(4)))  int    v4i;
typedef __attribute__((address_space(3)))    v4i    lds_v4i;

#if __has_builtin(__builtin_amdgcn_global_load_async_to_lds_b128)
#define HAVE_ASYNC_LDS 1
#else
#define HAVE_ASYNC_LDS 0
#endif

template <int N>
static __device__ __forceinline__ void async_wait() {
#if __has_builtin(__builtin_amdgcn_s_wait_asynccnt)
  __builtin_amdgcn_s_wait_asynccnt(N);
#else
  asm volatile("s_wait_asynccnt %0" ::"i"(N) : "memory");
#endif
}

static __device__ __forceinline__ unsigned short f2bf_bits(float f) {
  union { float f; unsigned u; } v; v.f = f;
  unsigned r = v.u + 0x7FFFu + ((v.u >> 16) & 1u);   // round-to-nearest-even
  return (unsigned short)(r >> 16);
}
static __device__ __forceinline__ __bf16 f2bf(float f) {
  unsigned short h = f2bf_bits(f);
  __bf16 b; __builtin_memcpy(&b, &h, 2); return b;
}

// ---------------------------------------------------------------------------
// Tiled bf16 WMMA GEMM: C[M,N] = A[M,K] * Bt[N,K]^T + bias (+ epilogue)
//   mode 0: Cf = f32 result
//   mode 1: Cb = bf16 result
//   mode 2: Cf = f32 result + resid
// A row-major [M,K] bf16, Bt row-major [N,K] bf16 (i.e. W transposed).
// Block: 256 threads = 8 waves; tile 128x128, BK=32; LDS double-buffered with
// GLOBAL_LOAD_ASYNC_TO_LDS_B128 (ASYNCcnt) when the toolchain exposes it.
// Wave grid 4(M) x 2(N): each wave computes 32x64 = 2x4 WMMA tiles.
// ---------------------------------------------------------------------------
#define TM 128
#define TN 128
#define TK 32

__global__ __launch_bounds__(256)
void gemm_bf16_wmma(const __bf16* __restrict__ A, const __bf16* __restrict__ Bt,
                    const float* __restrict__ bias,
                    float* __restrict__ Cf, __bf16* __restrict__ Cb,
                    const float* __restrict__ resid,
                    int M, int N, int K, int mode)
{
  __shared__ __align__(16) __bf16 As[2][TM * TK];   // [m][k]
  __shared__ __align__(16) __bf16 Bs[2][TN * TK];   // [n][k]

  const int tid  = threadIdx.x;
  const int lane = tid & 31;
  const int wv   = tid >> 5;
  const int wm   = wv & 3;       // 4 waves along M (32 rows each)
  const int wn   = wv >> 2;      // 2 waves along N (64 cols each)
  const i64 blockM = (i64)blockIdx.y * TM;
  const i64 blockN = (i64)blockIdx.x * TN;

  const v8f zero8 = {0.f,0.f,0.f,0.f,0.f,0.f,0.f,0.f};
  v8f acc[2][4];
#pragma unroll
  for (int i = 0; i < 2; ++i)
#pragma unroll
    for (int j = 0; j < 4; ++j) acc[i][j] = zero8;

  // global->LDS copy mapping: 2 threads per 32-element (64B) tile row
  const int ldRow = tid >> 1;          // 0..127
  const int ldOff = (tid & 1) << 4;    // 0 or 16 (bf16 elements)
  const __bf16* gA = A  + (blockM + ldRow) * (i64)K + ldOff;
  const __bf16* gB = Bt + (blockN + ldRow) * (i64)K + ldOff;
  const int ldsOff = ldRow * TK + ldOff;

  const int half = lane >> 4;          // ISA bf16 A/B fragment half select
  const int lr   = lane & 15;

  // per-k-step WMMA compute from an LDS buffer pair
  auto compute = [&](const __bf16* as, const __bf16* bs) {
    // A fragment (16x32): lanes 0-15 hold row M=lr, K {0..7,16..23};
    // lanes 16-31 hold row M=lr, K {8..15,24..31}  (ISA 7.12.2)
    v16bf afrag[2], bfrag[4];
#pragma unroll
    for (int am = 0; am < 2; ++am) {
      const __bf16* p = as + (wm * 32 + am * 16 + lr) * TK + half * 8;
      union { v16bf v; struct { v8bf lo, hi; } s; } u;
      u.s.lo = *(const v8bf*)(p);
      u.s.hi = *(const v8bf*)(p + 16);
      afrag[am] = u.v;
    }
    // B fragment (32x16): lane n<16 holds col n, K {0..7,16..23}; lane n+16 K {8..15,24..31}
#pragma unroll
    for (int bn = 0; bn < 4; ++bn) {
      const __bf16* p = bs + (wn * 64 + bn * 16 + lr) * TK + half * 8;
      union { v16bf v; struct { v8bf lo, hi; } s; } u;
      u.s.lo = *(const v8bf*)(p);
      u.s.hi = *(const v8bf*)(p + 16);
      bfrag[bn] = u.v;
    }
#pragma unroll
    for (int am = 0; am < 2; ++am)
#pragma unroll
      for (int bn = 0; bn < 4; ++bn)
        acc[am][bn] = __builtin_amdgcn_wmma_f32_16x16x32_bf16(
            false, afrag[am], false, bfrag[bn], (short)0, acc[am][bn],
            false, false);
  };

#if HAVE_ASYNC_LDS
  // ---- async double-buffered pipeline (VGPR-bypassing global->LDS DMA) ----
  auto issue = [&](int buf, int k0) {
    __builtin_amdgcn_global_load_async_to_lds_b128(
        (v4i*)(gA + k0), (lds_v4i*)(&As[buf][ldsOff]), 0, 0);
    __builtin_amdgcn_global_load_async_to_lds_b128(
        (v4i*)(gB + k0), (lds_v4i*)(&Bs[buf][ldsOff]), 0, 0);
  };
  const int nk = K / TK;
  issue(0, 0);
  for (int kt = 0; kt < nk; ++kt) {
    if (kt + 1 < nk) {
      issue((kt + 1) & 1, (kt + 1) * TK);  // overlap next stage with compute
      async_wait<2>();                     // this wave's stage-kt loads landed
    } else {
      async_wait<0>();
    }
    __syncthreads();                       // whole tile (all waves) in LDS
    compute(As[kt & 1], Bs[kt & 1]);
    __syncthreads();                       // done reading before overwrite
  }
#else
  // ---- fallback: synchronous staging through VGPRs ----
  for (int k0 = 0; k0 < K; k0 += TK) {
    const uint4 av = *(const uint4*)(gA + k0);
    const uint4 bv = *(const uint4*)(gB + k0);
    if (k0 + TK < K) {
      __builtin_prefetch(gA + k0 + TK, 0, 0);
      __builtin_prefetch(gB + k0 + TK, 0, 0);
    }
    __syncthreads();
    *(uint4*)(&As[0][ldsOff]) = av;
    *(uint4*)(&Bs[0][ldsOff]) = bv;
    __syncthreads();
    compute(As[0], Bs[0]);
  }
#endif

  // Epilogue. C/D layout: lane = n + 16*(m>=8), vgpr v -> m = 8*half + v.
#pragma unroll
  for (int am = 0; am < 2; ++am) {
    const i64 row0 = blockM + wm * 32 + am * 16 + half * 8;
#pragma unroll
    for (int bn = 0; bn < 4; ++bn) {
      const i64 col = blockN + wn * 64 + bn * 16 + lr;
      const float bb = bias[col];
#pragma unroll
      for (int v = 0; v < 8; ++v) {
        const i64 off = (row0 + v) * (i64)N + col;
        float val = acc[am][bn][v] + bb;
        if (mode == 2) val += resid[off];
        if (mode == 1) Cb[off] = f2bf(val);
        else           Cf[off] = val;
      }
    }
  }
}

// ---------------------------------------------------------------------------
// Weight transpose+convert: Wt[n][k] = bf16(W[k][n])
// ---------------------------------------------------------------------------
__global__ __launch_bounds__(256)
void wtrans_kernel(const float* __restrict__ W, __bf16* __restrict__ Wt,
                   int K, int N)
{
  __shared__ float tile[32][33];
  const int n0 = blockIdx.x * 32;
  const int k0 = blockIdx.y * 32;
  const int tx = threadIdx.x & 31;
  const int ty = threadIdx.x >> 5;
#pragma unroll
  for (int r = ty; r < 32; r += 8)
    tile[r][tx] = W[(i64)(k0 + r) * N + n0 + tx];
  __syncthreads();
#pragma unroll
  for (int r = ty; r < 32; r += 8)
    Wt[(i64)(n0 + r) * K + k0 + tx] = f2bf(tile[tx][r]);
}

// ---------------------------------------------------------------------------
// f32 -> bf16 bulk convert (4 elements/thread)
// ---------------------------------------------------------------------------
__global__ __launch_bounds__(256)
void cvt_bf16_kernel(const float* __restrict__ in, __bf16* __restrict__ out,
                     i64 n4)
{
  const i64 t = (i64)blockIdx.x * blockDim.x + threadIdx.x;
  if (t >= n4) return;
  const float4 v = ((const float4*)in)[t];
  ushort4 o;
  o.x = f2bf_bits(v.x); o.y = f2bf_bits(v.y);
  o.z = f2bf_bits(v.z); o.w = f2bf_bits(v.w);
  ((ushort4*)out)[t] = o;
}

// ---------------------------------------------------------------------------
// Phasor memory: per (chunk-row, 4 dims): running cumsum over chunk of
// value*cos(kp), value*sin(kp); retrieved = (mr*cos(qp)+mi*sin(qp))/sqrt(S).
// Overwrites vr (value) in place with retrieved — each element is read
// before being written within its own iteration.
// ---------------------------------------------------------------------------
__global__ __launch_bounds__(256)
void phasor_kernel(float* __restrict__ vr, const float* __restrict__ kp,
                   const float* __restrict__ qp, int rows, int D, int C,
                   float scale)
{
  const int dv = D >> 2;
  const i64 t = (i64)blockIdx.x * blockDim.x + threadIdx.x;
  if (t >= (i64)rows * dv) return;
  const int d4 = (int)(t % dv);
  const i64 cr = t / dv;
  const i64 base = cr * (i64)C * D + (i64)d4 * 4;
  float mr[4] = {0,0,0,0}, mi[4] = {0,0,0,0};
  for (int p = 0; p < C; ++p) {
    const i64 off = base + (i64)p * D;
    const float4 v = *(const float4*)(vr + off);
    const float4 k = *(const float4*)(kp + off);
    const float4 q = *(const float4*)(qp + off);
    const float vv[4] = {v.x, v.y, v.z, v.w};
    const float kk[4] = {k.x, k.y, k.z, k.w};
    const float qq[4] = {q.x, q.y, q.z, q.w};
    float4 o;
    float* po = &o.x;
#pragma unroll
    for (int i = 0; i < 4; ++i) {
      float kc, ks, qc, qs;
      __sincosf(kk[i], &ks, &kc);
      __sincosf(qq[i], &qs, &qc);
      mr[i] += vv[i] * kc;
      mi[i] += vv[i] * ks;
      po[i] = (mr[i] * qc + mi[i] * qs) * scale;
    }
    *(float4*)(vr + off) = o;
  }
}

// ---------------------------------------------------------------------------
// LayerNorm per token (D=1024, 256 threads -> 4 elems/thread), writes bf16
// ---------------------------------------------------------------------------
__global__ __launch_bounds__(256)
void ln_kernel(const float* __restrict__ x, const float* __restrict__ g,
               const float* __restrict__ bta, __bf16* __restrict__ out, int D)
{
  __shared__ float s1[256], s2[256];
  const i64 row = blockIdx.x;
  const int tid = threadIdx.x;
  const float4 v = *(const float4*)(x + row * D + tid * 4);
  s1[tid] = v.x + v.y + v.z + v.w;
  s2[tid] = v.x*v.x + v.y*v.y + v.z*v.z + v.w*v.w;
  __syncthreads();
  for (int s = 128; s > 0; s >>= 1) {
    if (tid < s) { s1[tid] += s1[tid + s]; s2[tid] += s2[tid + s]; }
    __syncthreads();
  }
  const float mu   = s1[0] / D;
  const float var  = s2[0] / D - mu * mu;
  const float rstd = rsqrtf(var + 1e-5f);
  const float pv[4] = {v.x, v.y, v.z, v.w};
  ushort4 o;
  unsigned short* po = &o.x;
#pragma unroll
  for (int i = 0; i < 4; ++i) {
    const int d = tid * 4 + i;
    po[i] = f2bf_bits((pv[i] - mu) * rstd * g[d] + bta[d]);
  }
  ((ushort4*)out)[row * (D >> 2) + tid] = o;
}

// ---------------------------------------------------------------------------
// Ortho loss: one block per (b,i,j); sum_d (cos(ai)cos(aj)+sin(ai)sin(aj))^2
// ---------------------------------------------------------------------------
__global__ __launch_bounds__(256)
void ortho_kernel(const float* __restrict__ kp, const int* __restrict__ idx,
                  int n, int S, int D, float invDenom, float* __restrict__ outv)
{
  __shared__ float red[256];
  const int j = blockIdx.x % n;
  const int i = (blockIdx.x / n) % n;
  const int b = blockIdx.x / (n * n);
  float acc = 0.f;
  if (i != j) {
    const float* ri = kp + ((i64)b * S + idx[i]) * (i64)D;
    const float* rj = kp + ((i64)b * S + idx[j]) * (i64)D;
    for (int d = threadIdx.x; d < D; d += 256) {
      float si, ci, sj, cj;
      __sincosf(ri[d], &si, &ci);
      __sincosf(rj[d], &sj, &cj);
      const float dot = ci * cj + si * sj;
      acc += dot * dot;
    }
  }
  red[threadIdx.x] = acc;
  __syncthreads();
  for (int s = 128; s > 0; s >>= 1) {
    if (threadIdx.x < s) red[threadIdx.x] += red[threadIdx.x + s];
    __syncthreads();
  }
  if (threadIdx.x == 0) atomicAdd(outv, red[0] * invDenom);
}

__global__ void zero1_kernel(float* p) {
  if (threadIdx.x == 0 && blockIdx.x == 0) *p = 0.f;
}

// ---------------------------------------------------------------------------
extern "C" void kernel_launch(void* const* d_in, const int* in_sizes, int n_in,
                              void* d_out, int out_size, void* d_ws, size_t ws_size,
                              hipStream_t stream)
{
  const float* x    = (const float*)d_in[0];
  const float* Wk   = (const float*)d_in[1];
  const float* bk   = (const float*)d_in[2];
  const float* Wv   = (const float*)d_in[3];
  const float* bv   = (const float*)d_in[4];
  const float* Wq   = (const float*)d_in[5];
  const float* bq   = (const float*)d_in[6];
  const float* Wkp  = (const float*)d_in[7];
  const float* bkp  = (const float*)d_in[8];
  const float* Wqp  = (const float*)d_in[9];
  const float* bqp  = (const float*)d_in[10];
  const float* ln_g = (const float*)d_in[11];
  const float* ln_b = (const float*)d_in[12];
  const float* Wo   = (const float*)d_in[13];
  const float* bo   = (const float*)d_in[14];
  const int*   idx  = (const int*)d_in[15];
  const int n = in_sizes[15];          // 32 ortho samples

  const int B = 4, S = 4096, D = 1024, CHUNK = 64;  // chunk_size per reference
  const i64 M  = (i64)B * S;           // 16384 tokens
  const i64 MD = M * D;                // 16,777,216 elements

  // Workspace carve-up (~332 MB)
  char* w = (char*)d_ws;
  __bf16* x_bf   = (__bf16*)w;  w += MD * 2;
  __bf16* Wt     = (__bf16*)w;  w += (i64)6 * D * D * 2;
  __bf16* key_bf = (__bf16*)w;  w += MD * 2;
  __bf16* qry_bf = (__bf16*)w;  w += MD * 2;
  float*  val_f  = (float*) w;  w += MD * 4;   // value -> retrieved (in place)
  float*  kph    = (float*) w;  w += MD * 4;   // key_phase f32
  float*  qph    = (float*) w;  w += MD * 4;   // query_phase f32
  __bf16* ln_bf  = (__bf16*)w;  w += MD * 2;

  float* out_f   = (float*)d_out;      // (B,S,D) f32
  float* ortho_f = out_f + MD;         // trailing scalar

  const dim3 b256(256);

  zero1_kernel<<<1, 32, 0, stream>>>(ortho_f);
  cvt_bf16_kernel<<<dim3((unsigned)((MD / 4 + 255) / 256)), b256, 0, stream>>>(x, x_bf, MD / 4);

  const dim3 tg(D / 32, D / 32);
  wtrans_kernel<<<tg, b256, 0, stream>>>(Wk,  Wt + (i64)0 * D * D, D, D);
  wtrans_kernel<<<tg, b256, 0, stream>>>(Wv,  Wt + (i64)1 * D * D, D, D);
  wtrans_kernel<<<tg, b256, 0, stream>>>(Wq,  Wt + (i64)2 * D * D, D, D);
  wtrans_kernel<<<tg, b256, 0, stream>>>(Wkp, Wt + (i64)3 * D * D, D, D);
  wtrans_kernel<<<tg, b256, 0, stream>>>(Wqp, Wt + (i64)4 * D * D, D, D);
  wtrans_kernel<<<tg, b256, 0, stream>>>(Wo,  Wt + (i64)5 * D * D, D, D);

  const dim3 gg((unsigned)(D / TN), (unsigned)(M / TM));   // (8, 128)
  // key = x@Wk  (bf16 out), value = x@Wv (f32 out), query = x@Wq (bf16 out)
  gemm_bf16_wmma<<<gg, b256, 0, stream>>>(x_bf, Wt + (i64)0*D*D, bk, nullptr, key_bf, nullptr, (int)M, D, D, 1);
  gemm_bf16_wmma<<<gg, b256, 0, stream>>>(x_bf, Wt + (i64)1*D*D, bv, val_f,  nullptr, nullptr, (int)M, D, D, 0);
  gemm_bf16_wmma<<<gg, b256, 0, stream>>>(x_bf, Wt + (i64)2*D*D, bq, nullptr, qry_bf, nullptr, (int)M, D, D, 1);
  // key_phase = key@Wkp, query_phase = query@Wqp (f32 out)
  gemm_bf16_wmma<<<gg, b256, 0, stream>>>(key_bf, Wt + (i64)3*D*D, bkp, kph, nullptr, nullptr, (int)M, D, D, 0);
  gemm_bf16_wmma<<<gg, b256, 0, stream>>>(qry_bf, Wt + (i64)4*D*D, bqp, qph, nullptr, nullptr, (int)M, D, D, 0);

  const float invDenom = 1.0f / ((float)n * (float)(n - 1) * (float)D + 1e-6f);
  ortho_kernel<<<dim3((unsigned)(B * n * n)), b256, 0, stream>>>(kph, idx, n, S, D, invDenom, ortho_f);

  const int rows = B * (S / CHUNK);                 // 256 chunk-rows
  const i64 pthreads = (i64)rows * (D / 4);
  phasor_kernel<<<dim3((unsigned)((pthreads + 255) / 256)), b256, 0, stream>>>(
      val_f, kph, qph, rows, D, CHUNK, 0.015625f /* 1/sqrt(4096) */);

  ln_kernel<<<dim3((unsigned)M), b256, 0, stream>>>(val_f, ln_g, ln_b, ln_bf, D);

  // out = x + ln@Wo + bo
  gemm_bf16_wmma<<<gg, b256, 0, stream>>>(ln_bf, Wt + (i64)5*D*D, bo, out_f, nullptr, x, (int)M, D, D, 2);
}
